// CausalSelfAttention_21784074125342
// MI455X (gfx1250) — matrix operand order repaired and checked by reading, not compiled, not run on hardware
//
#include <hip/hip_runtime.h>

typedef __attribute__((ext_vector_type(16))) __bf16 v16bf;
typedef __attribute__((ext_vector_type(8)))  float  v8f;
typedef __attribute__((ext_vector_type(4)))  unsigned int u32x4;
typedef __attribute__((ext_vector_type(8)))  int i32x8;
typedef __attribute__((ext_vector_type(4)))  int i32x4;

union Frag16 { uint4 u[2]; v16bf v; };
union Pack16 { uint4 u[2]; __bf16 e[16]; };

static __device__ __forceinline__ v8f zero8() {
    v8f z;
#pragma unroll
    for (int i = 0; i < 8; ++i) z[i] = 0.0f;
    return z;
}

// ---------------------------------------------------------------------------
// Elementwise f32 -> bf16 convert
// ---------------------------------------------------------------------------
__global__ void cvt_f32_bf16(const float* __restrict__ s, __bf16* __restrict__ d, int n) {
    int i = blockIdx.x * blockDim.x + threadIdx.x;
    int stride = gridDim.x * blockDim.x;
    for (; i < n; i += stride) d[i] = (__bf16)s[i];
}

// s: [K][N] f32 row-major  ->  d: [N][K] bf16 (i.e. transposed)
__global__ void cvt_transpose_bf16(const float* __restrict__ s, __bf16* __restrict__ d,
                                   int K, int N) {
    int i = blockIdx.x * blockDim.x + threadIdx.x;
    int total = K * N;
    int stride = gridDim.x * blockDim.x;
    for (; i < total; i += stride) {
        int n = i / K;
        int k = i - n * K;
        d[i] = (__bf16)s[(size_t)k * N + n];
    }
}

// ---------------------------------------------------------------------------
// bf16 WMMA GEMM:  C[M][N] = A[M][K] * B[K][N],  B given transposed as Bt[N][K]
// Block tile 128x128, K-step 32, 8 waves, each wave -> 32x64 (2x4 WMMA tiles).
// mode 0: scatter bf16 into Q/K/V [B=2][H=16][T=2048][64]  (N = 3072)
// mode 1: store f32 to fo[M][N]                            (N = 1024)
// ---------------------------------------------------------------------------
#define LDSP 40   // padded LDS row stride (elements): 80 B -> conflict-free b128 reads

__global__ void __launch_bounds__(256)
gemm_bf16(const __bf16* __restrict__ A, const __bf16* __restrict__ Bt,
          int M, int N, int K, int mode,
          __bf16* __restrict__ qo, __bf16* __restrict__ ko, __bf16* __restrict__ vo,
          float* __restrict__ fo)
{
    __shared__ __bf16 As[128 * LDSP];
    __shared__ __bf16 Bs[128 * LDSP];

    const int tid  = threadIdx.x;
    const int lane = tid & 31;
    const int wid  = tid >> 5;
    const int wm   = wid >> 1;      // 0..3  (M direction)
    const int wn   = wid & 1;       // 0..1  (N direction)
    const int r16  = lane & 15;
    const int kh   = lane >> 4;
    const int m0   = blockIdx.y * 128;
    const int n0   = blockIdx.x * 128;

    v8f acc[2][4];
#pragma unroll
    for (int mt = 0; mt < 2; ++mt)
#pragma unroll
        for (int nt = 0; nt < 4; ++nt) acc[mt][nt] = zero8();

    const int rowS  = tid >> 1;
    const int halfS = tid & 1;
    const __bf16* aSrc = A  + (size_t)(m0 + rowS) * K + halfS * 16;
    const __bf16* bSrc = Bt + (size_t)(n0 + rowS) * K + halfS * 16;
    __bf16* aDst = &As[rowS * LDSP + halfS * 16];
    __bf16* bDst = &Bs[rowS * LDSP + halfS * 16];

    for (int k0 = 0; k0 < K; k0 += 32) {
        __syncthreads();
        {
            const uint4* sa = (const uint4*)(aSrc + k0);
            ((uint4*)aDst)[0] = sa[0];
            ((uint4*)aDst)[1] = sa[1];
            const uint4* sb = (const uint4*)(bSrc + k0);
            ((uint4*)bDst)[0] = sb[0];
            ((uint4*)bDst)[1] = sb[1];
        }
        __syncthreads();

#pragma unroll
        for (int mt = 0; mt < 2; ++mt) {
            Frag16 af;
            const int row = wm * 32 + mt * 16 + r16;
            af.u[0] = *(const uint4*)&As[row * LDSP + kh * 8];
            af.u[1] = *(const uint4*)&As[row * LDSP + 16 + kh * 8];
#pragma unroll
            for (int nt = 0; nt < 4; ++nt) {
                Frag16 bf_;
                const int col = wn * 64 + nt * 16 + r16;
                bf_.u[0] = *(const uint4*)&Bs[col * LDSP + kh * 8];
                bf_.u[1] = *(const uint4*)&Bs[col * LDSP + 16 + kh * 8];
                acc[mt][nt] = __builtin_amdgcn_wmma_f32_16x16x32_bf16(
                    false, af.v, false, bf_.v, (short)0, acc[mt][nt], false, false);
            }
        }
    }

#pragma unroll
    for (int mt = 0; mt < 2; ++mt) {
#pragma unroll
        for (int nt = 0; nt < 4; ++nt) {
#pragma unroll
            for (int g = 0; g < 8; ++g) {
                const int gm = m0 + wm * 32 + mt * 16 + g + 8 * kh;
                const int gn = n0 + wn * 64 + nt * 16 + r16;
                const float v = acc[mt][nt][g];
                if (mode == 0) {
                    const int which = gn >> 10;
                    const int dcol  = gn & 1023;
                    const int h     = dcol >> 6;
                    const int hd    = dcol & 63;
                    const int b     = gm >> 11;
                    const int t     = gm & 2047;
                    const size_t idx = (((size_t)(b * 16 + h)) * 2048 + t) * 64 + hd;
                    const __bf16 bv = (__bf16)v;
                    if (which == 0)      qo[idx] = bv;
                    else if (which == 1) ko[idx] = bv;
                    else                 vo[idx] = bv;
                } else {
                    fo[(size_t)gm * N + gn] = v;
                }
            }
        }
    }
}

// ---------------------------------------------------------------------------
// TDM: DMA one K tile (32 keys x 64 dims, bf16) global -> LDS, padding each
// 128B row by 16B so LDS rows have a 72-element (144B) stride (conflict-free
// b128 fragment reads). 2-D tensor: descriptor groups 2/3 zero.
// ---------------------------------------------------------------------------
static __device__ __forceinline__ void tdm_load_k_tile(const __bf16* gsrc,
                                                       unsigned ldsByteOff) {
    const unsigned long long ga = (unsigned long long)(uintptr_t)gsrc;
    u32x4 g0;
    g0[0] = 1u;                                            // count=1, user mode
    g0[1] = ldsByteOff;                                    // lds_addr
    g0[2] = (unsigned)ga;                                  // global_addr[31:0]
    g0[3] = (unsigned)((ga >> 32) & 0x1FFFFFFu) | (2u << 30); // addr[56:32] | type=2
    i32x8 g1;
    g1[0] = (int)((1u << 16) |            // data_size = 1 -> 2 bytes
                  (1u << 20) |            // pad_enable
                  (4u << 22) |            // pad_interval: 2^4*8B = 128B
                  (3u << 25));            // pad_amount: 4 DWORDs = 16B
    g1[1] = (int)(64u << 16);             // tensor_dim0 = 64 (lo16 in [63:48])
    g1[2] = (int)(2048u << 16);           // tensor_dim1 = 2048 (lo16 in [95:80])
    g1[3] = (int)(64u << 16);             // tile_dim0 = 64 ([127:112])
    g1[4] = 32;                           // tile_dim1 = 32 ([143:128]), tile_dim2=0
    g1[5] = 64;                           // tensor_dim0_stride = 64
    g1[6] = 0;
    g1[7] = 0;
    const i32x4 z4 = {0, 0, 0, 0};
#if __clang_major__ >= 23
    const i32x8 z8 = {0, 0, 0, 0, 0, 0, 0, 0};
    __builtin_amdgcn_tensor_load_to_lds(g0, g1, z4, z4, z8, 0);
#else
    __builtin_amdgcn_tensor_load_to_lds(g0, g1, z4, z4, 0);
#endif
}

// ---------------------------------------------------------------------------
// Flash attention forward. 128-thread block = 4 waves handling 64 consecutive
// q rows of one (b,h); K/V tiles (32 keys) shared through LDS:
//   K tile via TDM tensor_load_to_lds (wave 0 issues, s_wait_tensorcnt)
//   V tile transposed to Vt[dim][key] by all threads (b128 in, b16 scatter)
// Q/K/V: bf16 [B*H][T][64].  Output y: bf16 [B*T][1024] (token-major).
// ---------------------------------------------------------------------------
#define KSP 72   // Ks row stride (elements) = 64 + TDM pad -> conflict-free
#define VTP 40   // Vt row stride (elements)

__global__ void __launch_bounds__(128)
attn_fwd(const __bf16* __restrict__ qb, const __bf16* __restrict__ kb,
         const __bf16* __restrict__ vb, __bf16* __restrict__ yb)
{
    __shared__ __bf16 Ks[32 * KSP];     // [key][dim], 144B row stride
    __shared__ __bf16 Vt[64 * VTP];     // [dim][key], 80B row stride
    __shared__ __bf16 Pt[4][16 * 32];   // per-wave P bounce tile

    const int tid   = threadIdx.x;
    const int lane  = tid & 31;
    const int wid   = tid >> 5;
    const int r16   = lane & 15;
    const int kh    = lane >> 4;
    const int bh    = blockIdx.x >> 5;        // 32 blocks of 64 q-rows per head
    const int qBase0 = (blockIdx.x & 31) * 64;
    const int qBase  = qBase0 + wid * 16;     // this wave's 16 q rows
    const int b     = bh >> 4;
    const int h     = bh & 15;
    const float sc  = 1.44269504088896340736f * 0.125f;  // log2(e)/sqrt(64)

    // Q fragments (A layout): lane row r16, K groups {kh*8..} and {16+kh*8..}
    const __bf16* qrow = qb + ((size_t)bh * 2048 + qBase + r16) * 64;
    Frag16 qf[2];
#pragma unroll
    for (int kd = 0; kd < 2; ++kd) {
        qf[kd].u[0] = *(const uint4*)(qrow + kd * 32 + kh * 8);
        qf[kd].u[1] = *(const uint4*)(qrow + kd * 32 + 16 + kh * 8);
    }

    v8f acc[4];
#pragma unroll
    for (int nt = 0; nt < 4; ++nt) acc[nt] = zero8();
    float m8[8], l8[8];
#pragma unroll
    for (int g = 0; g < 8; ++g) { m8[g] = -1.0e30f; l8[g] = 0.0f; }

    const unsigned ksOff = (unsigned)(uintptr_t)(&Ks[0]);
    const int vkey = tid >> 2;          // 0..31 : staging assignment for V
    const int vseg = tid & 3;           // 16-dim segment

    const int nKB = (qBase0 + 64 + 31) >> 5;   // 32-key blocks for whole block

    for (int kbi = 0; kbi < nKB; ++kbi) {
        const int kBase = kbi * 32;

        // ---- stage K tile via TDM (wave 0 only; uniform descriptor)
        if (wid == 0)
            tdm_load_k_tile(kb + ((size_t)bh * 2048 + kBase) * 64, ksOff);

        // ---- stage V tile transposed: Vt[dim][key]
        {
            const __bf16* vsrc = vb + ((size_t)bh * 2048 + kBase + vkey) * 64 + vseg * 16;
            Pack16 w;
            w.u[0] = ((const uint4*)vsrc)[0];
            w.u[1] = ((const uint4*)vsrc)[1];
#pragma unroll
            for (int i = 0; i < 16; ++i)
                Vt[(vseg * 16 + i) * VTP + vkey] = w.e[i];
        }

        if (wid == 0) __builtin_amdgcn_s_wait_tensorcnt(0);
        __syncthreads();

        // ---- S = Q * K^T (scaled, causal-masked), base-2 log domain
        float sl[2][8];
#pragma unroll
        for (int nb = 0; nb < 2; ++nb) {
            v8f s = zero8();
#pragma unroll
            for (int kd = 0; kd < 2; ++kd) {
                // B fragment: lane col = key (r16), K = dims {kh*8..} and {16+kh*8..}
                Frag16 kf;
                const __bf16* krow = &Ks[(nb * 16 + r16) * KSP + kd * 32 + kh * 8];
                kf.u[0] = *(const uint4*)(krow);
                kf.u[1] = *(const uint4*)(krow + 16);
                s = __builtin_amdgcn_wmma_f32_16x16x32_bf16(
                    false, qf[kd].v, false, kf.v, (short)0, s, false, false);
            }
#pragma unroll
            for (int g = 0; g < 8; ++g) {
                const int qr = qBase + g + 8 * kh;
                const int ky = kBase + nb * 16 + r16;
                sl[nb][g] = (ky <= qr) ? s[g] * sc : -1.0e30f;
            }
        }

        // ---- online softmax (rows per half-wave; reduce across 16 lanes)
        float alpha[8];
#pragma unroll
        for (int g = 0; g < 8; ++g) {
            float mx = fmaxf(sl[0][g], sl[1][g]);
#pragma unroll
            for (int msk = 8; msk >= 1; msk >>= 1)
                mx = fmaxf(mx, __shfl_xor(mx, msk, 32));
            const float mnew = fmaxf(m8[g], mx);
            alpha[g] = exp2f(m8[g] - mnew);
            m8[g] = mnew;
            const float p0 = exp2f(sl[0][g] - mnew);
            const float p1 = exp2f(sl[1][g] - mnew);
            sl[0][g] = p0;
            sl[1][g] = p1;
            float rs = p0 + p1;
#pragma unroll
            for (int msk = 8; msk >= 1; msk >>= 1)
                rs += __shfl_xor(rs, msk, 32);
            l8[g] = l8[g] * alpha[g] + rs;
        }
#pragma unroll
        for (int nt = 0; nt < 4; ++nt)
#pragma unroll
            for (int g = 0; g < 8; ++g) acc[nt][g] *= alpha[g];

        // ---- bounce P through LDS: D layout -> A layout (same-wave, in-order)
        __bf16* pw = &Pt[wid][0];
#pragma unroll
        for (int nb = 0; nb < 2; ++nb)
#pragma unroll
            for (int g = 0; g < 8; ++g)
                pw[(g + 8 * kh) * 32 + nb * 16 + r16] = (__bf16)sl[nb][g];
        Frag16 pf;
#pragma unroll
        for (int j = 0; j < 8; ++j) {
            pf.v[j]     = pw[r16 * 32 + kh * 8 + j];
            pf.v[8 + j] = pw[r16 * 32 + 16 + kh * 8 + j];
        }

        // ---- acc += P * V  (V B-fragment from Vt: col = dim, K = keys kh*8+j, 16+kh*8+j)
#pragma unroll
        for (int nt = 0; nt < 4; ++nt) {
            Frag16 vf;
            const __bf16* vrow = &Vt[(nt * 16 + r16) * VTP + kh * 8];
            vf.u[0] = *(const uint4*)(vrow);
            vf.u[1] = *(const uint4*)(vrow + 16);
            acc[nt] = __builtin_amdgcn_wmma_f32_16x16x32_bf16(
                false, pf.v, false, vf.v, (short)0, acc[nt], false, false);
        }

        __syncthreads();   // protect Ks/Vt before next stage
    }

    // ---- normalize and store y in token-major [B*T][1024] bf16 for GEMM2
#pragma unroll
    for (int nt = 0; nt < 4; ++nt) {
#pragma unroll
        for (int g = 0; g < 8; ++g) {
            const int row = qBase + g + 8 * kh;
            const float v = acc[nt][g] / l8[g];
            yb[((size_t)(b * 2048) + row) * 1024 + h * 64 + nt * 16 + r16] = (__bf16)v;
        }
    }
}

// ---------------------------------------------------------------------------
// Host-side orchestration
// ---------------------------------------------------------------------------
extern "C" void kernel_launch(void* const* d_in, const int* in_sizes, int n_in,
                              void* d_out, int out_size, void* d_ws, size_t ws_size,
                              hipStream_t stream) {
    const float* x     = (const float*)d_in[0];   // [2,2048,1024]
    const float* Wqkv  = (const float*)d_in[1];   // [1024,3072]
    const float* Wproj = (const float*)d_in[2];   // [1024,1024]
    float* out = (float*)d_out;                   // [2,2048,1024]

    char* ws = (char*)d_ws;
    __bf16* xb     = (__bf16*)(ws);                          // 4096x1024      (8 MB)
    __bf16* wqkvT  = (__bf16*)(ws + ((size_t)8  << 20));     // 3072x1024 (W^T, 6 MB)
    __bf16* wprojT = (__bf16*)(ws + ((size_t)14 << 20));     // 1024x1024 (W^T, 2 MB)
    __bf16* qbuf   = (__bf16*)(ws + ((size_t)16 << 20));     // [32][2048][64] (8 MB)
    __bf16* kbuf   = (__bf16*)(ws + ((size_t)24 << 20));     // [32][2048][64] (8 MB)
    __bf16* vbuf   = (__bf16*)(ws + ((size_t)32 << 20));     // [32][2048][64] (8 MB)
    __bf16* ybuf   = (__bf16*)(ws + ((size_t)40 << 20));     // 4096x1024      (8 MB)

    cvt_f32_bf16<<<2048, 256, 0, stream>>>(x, xb, 4096 * 1024);
    cvt_transpose_bf16<<<2048, 256, 0, stream>>>(Wqkv, wqkvT, 1024, 3072);
    cvt_transpose_bf16<<<1024, 256, 0, stream>>>(Wproj, wprojT, 1024, 1024);

    gemm_bf16<<<dim3(24, 32), 256, 0, stream>>>(xb, wqkvT, 4096, 3072, 1024, 0,
                                                qbuf, kbuf, vbuf, nullptr);

    attn_fwd<<<1024, 128, 0, stream>>>(qbuf, kbuf, vbuf, ybuf);

    gemm_bf16<<<dim3(8, 32), 256, 0, stream>>>(ybuf, wprojT, 4096, 1024, 1024, 1,
                                               nullptr, nullptr, nullptr, out);
}
